// BVHRouterAdapter_62474594287686
// MI455X (gfx1250) — compile-verified, matrix operand order checked by reading, not run on hardware
//
#include <hip/hip_runtime.h>
#include <hip/hip_bf16.h>
#include <stdint.h>

#define N_TOKENS_   16384
#define HIDDEN_     2048
#define N_EXPERTS_  128
#define N_CAND_     32
#define TOPK_       2
#define EPS_        1e-6f

#define TTILE_      32          // tokens per workgroup
#define KTILE_      32          // K per WMMA
#define ASTRIDE_    4112        // LDS row stride in bytes: 2048*2 + 16 pad (16B aligned)

typedef __attribute__((ext_vector_type(16))) __bf16 v16bf;
typedef __attribute__((ext_vector_type(8)))  float  v8f;

union FragU { uint4 q[2]; v16bf v; };

__device__ __forceinline__ unsigned short f32_to_bf16_bits(float f) {
    union { float f; uint32_t u; } c; c.f = f;
    uint32_t u = c.u;
    uint32_t r = u + 0x7FFFu + ((u >> 16) & 1u);       // round-to-nearest-even
    if ((u & 0x7F800000u) == 0x7F800000u) r = u;        // keep inf/NaN payload
    return (unsigned short)(r >> 16);
}

// ---------------------------------------------------------------------------
// Kernel 1: convert both weight matrices f32 -> bf16 into workspace.
// Layout in d_ws: [mat(2)][expert(128)][hidden(2048)] bf16.
// ---------------------------------------------------------------------------
__global__ void wcvt_bf16_kernel(const float* __restrict__ proj_w,
                                 const float* __restrict__ bvh_w,
                                 unsigned short* __restrict__ out) {
    int i = blockIdx.x * blockDim.x + threadIdx.x;
    const int n = N_EXPERTS_ * HIDDEN_;
    if (i < n) {
        out[i]     = f32_to_bf16_bits(proj_w[i]);
        out[n + i] = f32_to_bf16_bits(bvh_w[i]);
    }
}

// ---------------------------------------------------------------------------
// Kernel 2: fused RMSNorm + dual WMMA GEMM + softmax + hybrid top-k router.
// 256 threads = 8 waves (wave32). 32 tokens per block.
// ---------------------------------------------------------------------------
__global__ __launch_bounds__(256)
void router_fused_kernel(const float* __restrict__ hs,
                         const float* __restrict__ norm_w,
                         const float* __restrict__ scale_p,
                         const float* __restrict__ root_p,
                         const float* __restrict__ proj_b,
                         const float* __restrict__ bvh_b,
                         const unsigned short* __restrict__ wbf16,
                         const float* __restrict__ pes,
                         float* __restrict__ out) {
    extern __shared__ char smem[];

    const int tid  = threadIdx.x;
    const int lane = tid & 31;
    const int wv   = tid >> 5;
    const int tok0 = blockIdx.x * TTILE_;

    const float sc = scale_p[0] * root_p[0];

    // ---------------- Phase 1: RMSNorm -> bf16 x tile in LDS ----------------
    for (int t = 0; t < 4; ++t) {
        const int tl = wv * 4 + t;
        const float* row = hs + (size_t)(tok0 + tl) * HIDDEN_;
        float ss = 0.f;
        for (int h = lane; h < HIDDEN_; h += 32) { float v = row[h]; ss += v * v; }
        for (int off = 16; off >= 1; off >>= 1) ss += __shfl_xor(ss, off, 32);
        const float rn = rsqrtf(ss * (1.0f / HIDDEN_) + EPS_) * sc;
        unsigned short* dst = (unsigned short*)(smem + (size_t)tl * ASTRIDE_);
        for (int h = lane; h < HIDDEN_; h += 32)
            dst[h] = f32_to_bf16_bits(row[h] * rn * norm_w[h]);
    }
    __syncthreads();

    // ---------------- Phase 2: dual GEMM via v_wmma_f32_16x16x32_bf16 -------
    // wave wv: mtile = wv>>2 ; 4 n-tiles = (wv&3)*4+i ; ntile 0..7 proj, 8..15 bvh
    const int mtile = wv >> 2;
    const int nt0   = (wv & 3) * 4;

    v8f acc[4];
    {
        v8f zero = {0.f, 0.f, 0.f, 0.f, 0.f, 0.f, 0.f, 0.f};
        for (int i = 0; i < 4; ++i) acc[i] = zero;
    }

    // A-fragment addressing per ISA 16-bit A 16x32 layout:
    // lanes 0-15: M=lane,    K = 0..7  and 16..23 (within the 32-K tile)
    // lanes16-31: M=lane-16, K = 8..15 and 24..31
    const int arow = mtile * 16 + (lane & 15);
    const int koff = (lane >> 4) * 8;                      // elements
    const char* aBase = smem + (size_t)arow * ASTRIDE_ + (size_t)koff * 2;

    // B-fragment: lane supplies N=expert, same K pattern; weights are [E][H]
    const unsigned short* bBase[4];
    for (int i = 0; i < 4; ++i) {
        const int nt  = nt0 + i;
        const int mat = nt >> 3;
        const int e   = (nt & 7) * 16 + (lane & 15);
        bBase[i] = wbf16 + ((size_t)mat * N_EXPERTS_ + e) * HIDDEN_ + koff;
    }

    for (int kt = 0; kt < HIDDEN_ / KTILE_; ++kt) {
        FragU a;
        const char* ap = aBase + (size_t)kt * (KTILE_ * 2);
        a.q[0] = *(const uint4*)(ap);
        a.q[1] = *(const uint4*)(ap + 32);                 // K += 16 elements
#pragma unroll
        for (int i = 0; i < 4; ++i) {
            FragU b;
            const unsigned short* bp = bBase[i] + (size_t)kt * KTILE_;
            b.q[0] = *(const uint4*)(bp);
            b.q[1] = *(const uint4*)(bp + 16);             // K += 16 elements
            acc[i] = __builtin_amdgcn_wmma_f32_16x16x32_bf16(
                false, a.v, false, b.v, (short)0, acc[i], false, false);
        }
    }

    __syncthreads();   // x tile dead; reuse its LDS for logits

    // ---------------- Phase 3: logits (+bias) -> LDS ------------------------
    float* lg = (float*)smem;                              // [2][TTILE_][128]
    for (int i = 0; i < 4; ++i) {
        const int nt  = nt0 + i;
        const int mat = nt >> 3;
        const int e   = (nt & 7) * 16 + (lane & 15);
        const float bias = (mat ? bvh_b : proj_b)[e];
#pragma unroll
        for (int r = 0; r < 8; ++r) {
            // C/D layout: VGPR r -> M = r (lanes 0-15) or 8+r (lanes 16-31)
            const int tl = mtile * 16 + r + ((lane >> 4) * 8);
            lg[((size_t)mat * TTILE_ + tl) * N_EXPERTS_ + e] = acc[i][r] + bias;
        }
    }
    __syncthreads();

    // ---------------- Phase 4: softmax + hybrid top-k -----------------------
    float* lgP = lg;
    float* lgB = lg + (size_t)TTILE_ * N_EXPERTS_;
    for (int t = 0; t < 4; ++t) {
        const int tl = wv * 4 + t;
        const int tg = tok0 + tl;

        float pl[4], bl[4];
#pragma unroll
        for (int j = 0; j < 4; ++j) {
            pl[j] = lgP[(size_t)tl * N_EXPERTS_ + lane + 32 * j];
            bl[j] = lgB[(size_t)tl * N_EXPERTS_ + lane + 32 * j];
        }

        // softmax over full proj logits (128 = 4 per lane)
        float m = fmaxf(fmaxf(pl[0], pl[1]), fmaxf(pl[2], pl[3]));
        for (int off = 16; off >= 1; off >>= 1) m = fmaxf(m, __shfl_xor(m, off, 32));
        float p[4], s = 0.f;
#pragma unroll
        for (int j = 0; j < 4; ++j) { p[j] = __expf(pl[j] - m); s += p[j]; }
        for (int off = 16; off >= 1; off >>= 1) s += __shfl_xor(s, off, 32);
        const float inv = 1.0f / s;
#pragma unroll
        for (int j = 0; j < 4; ++j) {
            p[j] *= inv;
            out[(size_t)tg * N_EXPERTS_ + lane + 32 * j] = p[j];
        }

        // top-32 candidate mask from bvh logits: argmax-and-remove, tie -> low index
        bool cand[4] = {false, false, false, false};
        float bv[4];
#pragma unroll
        for (int j = 0; j < 4; ++j) bv[j] = bl[j];
        for (int c = 0; c < N_CAND_; ++c) {
            float v = -3.4e38f; int idx = lane;
#pragma unroll
            for (int j = 0; j < 4; ++j)
                if (bv[j] > v) { v = bv[j]; idx = lane + 32 * j; }
            for (int off = 16; off >= 1; off >>= 1) {
                float ov = __shfl_xor(v, off, 32);
                int   oi = __shfl_xor(idx, off, 32);
                if (ov > v || (ov == v && oi < idx)) { v = ov; idx = oi; }
            }
            if ((idx & 31) == lane) { cand[idx >> 5] = true; bv[idx >> 5] = -3.4e38f; }
        }

        // top-2 of candidate-masked probs (probs >= 0, so -1/-2 are sentinels)
        float cv[4];
#pragma unroll
        for (int j = 0; j < 4; ++j) cv[j] = cand[j] ? p[j] : -1.0f;
        float tv[TOPK_]; int ti[TOPK_];
        for (int k = 0; k < TOPK_; ++k) {
            float v = -2.0f; int idx = lane;
#pragma unroll
            for (int j = 0; j < 4; ++j)
                if (cv[j] > v) { v = cv[j]; idx = lane + 32 * j; }
            for (int off = 16; off >= 1; off >>= 1) {
                float ov = __shfl_xor(v, off, 32);
                int   oi = __shfl_xor(idx, off, 32);
                if (ov > v || (ov == v && oi < idx)) { v = ov; idx = oi; }
            }
            tv[k] = v; ti[k] = idx;
            if ((idx & 31) == lane) cv[idx >> 5] = -2.0f;
        }

        if (lane == 0) {
            const float ssum = tv[0] + tv[1];
            const float w0 = tv[0] / ssum * pes[ti[0]];
            const float w1 = tv[1] / ssum * pes[ti[1]];
            float* outW = out + (size_t)N_TOKENS_ * N_EXPERTS_;
            float* outI = outW + (size_t)N_TOKENS_ * TOPK_;
            outW[(size_t)tg * 2 + 0] = w0;
            outW[(size_t)tg * 2 + 1] = w1;
            outI[(size_t)tg * 2 + 0] = (float)ti[0];
            outI[(size_t)tg * 2 + 1] = (float)ti[1];
        }
    }
}

// ---------------------------------------------------------------------------
extern "C" void kernel_launch(void* const* d_in, const int* in_sizes, int n_in,
                              void* d_out, int out_size, void* d_ws, size_t ws_size,
                              hipStream_t stream) {
    (void)in_sizes; (void)n_in; (void)out_size; (void)ws_size;

    const float* hs     = (const float*)d_in[0];
    const float* norm_w = (const float*)d_in[1];
    const float* scale  = (const float*)d_in[2];
    const float* root   = (const float*)d_in[3];
    const float* proj_w = (const float*)d_in[4];
    const float* proj_b = (const float*)d_in[5];
    const float* bvh_w  = (const float*)d_in[6];
    const float* bvh_b  = (const float*)d_in[7];
    const float* pes    = (const float*)d_in[8];
    float* out          = (float*)d_out;
    unsigned short* wbf = (unsigned short*)d_ws;   // 2*128*2048 bf16 = 1 MB

    const int nW = N_EXPERTS_ * HIDDEN_;
    wcvt_bf16_kernel<<<(nW + 255) / 256, 256, 0, stream>>>(proj_w, bvh_w, wbf);

    const size_t shmem = (size_t)TTILE_ * ASTRIDE_;       // 131584 B (> logits 32 KB)
    router_fused_kernel<<<N_TOKENS_ / TTILE_, 256, shmem, stream>>>(
        hs, norm_w, scale, root, proj_b, bvh_b, wbf, pes, out);
}
